// LowRankTriLinearFusionAttn_82609400971793
// MI455X (gfx1250) — matrix operand order repaired and checked by reading, not compiled, not run on hardware
//
#include <hip/hip_runtime.h>
#include <hip/hip_bf16.h>

// ---------------------------------------------------------------------------
// Types for CDNA5 WMMA (wave32): V_WMMA_F32_16X16X32_BF16
// ---------------------------------------------------------------------------
typedef __bf16 bf16_t;
typedef __attribute__((ext_vector_type(16))) __bf16 v16bf;
typedef __attribute__((ext_vector_type(8)))  __bf16 v8bf;
typedef __attribute__((ext_vector_type(8)))  float  v8f;
typedef __attribute__((ext_vector_type(4)))  float  v4f;

#define D_G   512
#define D_2D  768
#define D_3D  1024
#define D_F   256
#define RANK  16
#define ATTN_H 512
#define BATCH 8192
#define D_CAT (D_G + D_2D + D_3D)   // 2304

__device__ __forceinline__ v8f wmma_bf16(v16bf a, v16bf b, v8f c) {
  // 8 args: (neg_a, A, neg_b, B, c_mod, C, reuse_a, reuse_b)
  return __builtin_amdgcn_wmma_f32_16x16x32_bf16(false, a, false, b, (short)0, c, false, false);
}

// A-fragment (16x32, bf16) from row-major f32 X (ld = row stride).
// Lane L<16: row m0+L, K = k0..k0+7 and k0+16..k0+23
// Lane L>=16: row m0+(L-16), K = k0+8..k0+15 and k0+24..k0+31
__device__ __forceinline__ v16bf loadA_f32(const float* __restrict__ X, int ld,
                                           int m0, int k0, int lane) {
  int row = m0 + (lane & 15);
  int off = (lane < 16) ? 0 : 8;
  const float* p = X + (size_t)row * ld + k0 + off;
  const v4f* q0 = (const v4f*)p;
  const v4f* q1 = (const v4f*)(p + 16);
  v4f x0 = q0[0], x1 = q0[1], x2 = q1[0], x3 = q1[1];
  v16bf r;
#pragma unroll
  for (int i = 0; i < 4; ++i) {
    r[i]      = (__bf16)x0[i];
    r[4 + i]  = (__bf16)x1[i];
    r[8 + i]  = (__bf16)x2[i];
    r[12 + i] = (__bf16)x3[i];
  }
  return r;
}

// A-fragment from row-major bf16 memory (global or LDS pointer).
__device__ __forceinline__ v16bf loadA_bf16(const bf16_t* __restrict__ X, int ld,
                                            int m0, int k0, int lane) {
  int row = m0 + (lane & 15);
  int off = (lane < 16) ? 0 : 8;
  const bf16_t* p = X + (size_t)row * ld + k0 + off;
  v8bf lo = *(const v8bf*)p;
  v8bf hi = *(const v8bf*)(p + 16);
  return __builtin_shufflevector(lo, hi, 0, 1, 2, 3, 4, 5, 6, 7,
                                 8, 9, 10, 11, 12, 13, 14, 15);
}

// B-fragment (32x16, bf16) for y = x @ W^T : B[k][n] = W[n][k], W row-major (N,K).
// Lane L<16: col n0+L, K = k0..k0+7 and k0+16..k0+23 ; lane L>=16: col n0+(L-16), K = k0+8..k0+15 and k0+24..k0+31
__device__ __forceinline__ v16bf loadB_bf16(const bf16_t* __restrict__ W, int ldk,
                                            int n0, int k0, int lane) {
  int n  = n0 + (lane & 15);
  int kk = k0 + ((lane < 16) ? 0 : 8);
  const bf16_t* p = W + (size_t)n * ldk + kk;
  v8bf lo = *(const v8bf*)p;
  v8bf hi = *(const v8bf*)(p + 16);
  return __builtin_shufflevector(lo, hi, 0, 1, 2, 3, 4, 5, 6, 7,
                                 8, 9, 10, 11, 12, 13, 14, 15);
}

// ---------------------------------------------------------------------------
// Kernel 0: f32 -> bf16 conversion (all weights), one element per thread.
// ---------------------------------------------------------------------------
__global__ void k_cvt(const float* __restrict__ src, bf16_t* __restrict__ dst, int n) {
  int i = blockIdx.x * blockDim.x + threadIdx.x;
  if (i < n) dst[i] = (bf16_t)src[i];
}

// ---------------------------------------------------------------------------
// Kernel 1: y = relu(LN(X @ W^T + bias)) -> bf16
// 64 rows/block, 512 threads (16 waves): wave owns a 16-col strip x 4 row tiles;
// one B fragment feeds 4 WMMAs (A reuse of B halves fragment traffic).
// ---------------------------------------------------------------------------
__global__ __launch_bounds__(512)
void k_proj_ln_relu(const float* __restrict__ X, const bf16_t* __restrict__ Wb,
                    const float* __restrict__ bias,
                    const float* __restrict__ ln_w, const float* __restrict__ ln_b,
                    bf16_t* __restrict__ out, int D) {
  __shared__ float smem[64 * 256];      // 64 KB (of 320 KB WGP LDS)
  __shared__ float red0[64][8];
  __shared__ float red1[64][8];
  __shared__ float stats[64][2];

  const int m0   = blockIdx.x * 64;
  const int wave = threadIdx.x >> 5;   // 0..15
  const int lane = threadIdx.x & 31;
  const int nb   = wave * 16;

  v8f acc[4] = {};
  for (int k0 = 0; k0 < D; k0 += 32) {
    v16bf b = loadB_bf16(Wb, D, nb, k0, lane);
#pragma unroll
    for (int mt = 0; mt < 4; ++mt) {
      v16bf a = loadA_f32(X, D, m0 + mt * 16, k0, lane);
      acc[mt] = wmma_bf16(a, b, acc[mt]);
    }
  }

  const int n0 = nb + (lane & 15);
  const float bv = bias[n0];
#pragma unroll
  for (int mt = 0; mt < 4; ++mt) {
    const int rbase = mt * 16 + ((lane < 16) ? 0 : 8);
#pragma unroll
    for (int j = 0; j < 8; ++j)
      smem[(rbase + j) * 256 + n0] = acc[mt][j] + bv;
  }
  __syncthreads();

  // LayerNorm: row r = t>>3 (0..63), segment seg = t&7 covers 32 cols.
  const int r   = threadIdx.x >> 3;
  const int seg = threadIdx.x & 7;
  float s = 0.f, ss = 0.f;
#pragma unroll
  for (int i = 0; i < 32; ++i) {
    float v = smem[r * 256 + seg * 32 + i];
    s += v; ss += v * v;
  }
  red0[r][seg] = s; red1[r][seg] = ss;
  __syncthreads();
  if (threadIdx.x < 64) {
    float s2 = 0.f, ss2 = 0.f;
#pragma unroll
    for (int i = 0; i < 8; ++i) { s2 += red0[threadIdx.x][i]; ss2 += red1[threadIdx.x][i]; }
    float mu  = s2 * (1.f / 256.f);
    float var = ss2 * (1.f / 256.f) - mu * mu;   // biased variance (jnp.var)
    stats[threadIdx.x][0] = mu;
    stats[threadIdx.x][1] = rsqrtf(var + 1e-5f);
  }
  __syncthreads();
  const float mu = stats[r][0], rs = stats[r][1];
#pragma unroll
  for (int i = 0; i < 32; ++i) {
    int n = seg * 32 + i;
    float v = (smem[r * 256 + n] - mu) * rs * ln_w[n] + ln_b[n];
    out[(size_t)(m0 + r) * 256 + n] = (bf16_t)fmaxf(v, 0.f);
  }
}

// ---------------------------------------------------------------------------
// Kernel 2: hidden = relu(h_cat @ Wa1^T + ba1) -> bf16 (B x 512)
// 64 rows x 256 cols per block (512 threads); K tile picks the concat segment
// (segment boundaries 512 / 1280 are multiples of the 32-wide K tile).
// ---------------------------------------------------------------------------
__global__ __launch_bounds__(512)
void k_attn1(const float* __restrict__ hg, const float* __restrict__ h2,
             const float* __restrict__ h3, const bf16_t* __restrict__ Wa1b,
             const float* __restrict__ ba1, bf16_t* __restrict__ hidden) {
  const int m0   = blockIdx.x * 64;
  const int wave = threadIdx.x >> 5;
  const int lane = threadIdx.x & 31;
  const int nb   = blockIdx.y * 256 + wave * 16;

  v8f acc[4] = {};
  for (int k0 = 0; k0 < D_CAT; k0 += 32) {
    const float* X; int ld, kk;
    if (k0 < D_G)             { X = hg; ld = D_G;  kk = k0; }
    else if (k0 < D_G + D_2D) { X = h2; ld = D_2D; kk = k0 - D_G; }
    else                      { X = h3; ld = D_3D; kk = k0 - D_G - D_2D; }
    v16bf b = loadB_bf16(Wa1b, D_CAT, nb, k0, lane);
#pragma unroll
    for (int mt = 0; mt < 4; ++mt) {
      v16bf a = loadA_f32(X, ld, m0 + mt * 16, kk, lane);
      acc[mt] = wmma_bf16(a, b, acc[mt]);
    }
  }
  const int n = nb + (lane & 15);
  const float bv = ba1[n];
#pragma unroll
  for (int mt = 0; mt < 4; ++mt) {
    const int rbase = mt * 16 + ((lane < 16) ? 0 : 8);
#pragma unroll
    for (int j = 0; j < 8; ++j) {
      float v = fmaxf(acc[mt][j] + bv, 0.f);
      hidden[(size_t)(m0 + rbase + j) * ATTN_H + n] = (bf16_t)v;
    }
  }
}

// ---------------------------------------------------------------------------
// Kernel 3: beta = softmax(hidden @ Wa2^T + ba2) over RANK=16 (one wave / 16 rows)
// ---------------------------------------------------------------------------
__global__ __launch_bounds__(32)
void k_attn2_softmax(const bf16_t* __restrict__ hidden, const bf16_t* __restrict__ Wa2b,
                     const float* __restrict__ ba2, float* __restrict__ beta) {
  __shared__ float sm[16 * 16];
  const int m0   = blockIdx.x * 16;
  const int lane = threadIdx.x & 31;

  v8f acc = {};
#pragma unroll
  for (int k0 = 0; k0 < ATTN_H; k0 += 32) {
    v16bf a = loadA_bf16(hidden, ATTN_H, m0, k0, lane);
    v16bf b = loadB_bf16(Wa2b, ATTN_H, 0, k0, lane);
    acc = wmma_bf16(a, b, acc);
  }
  const int n = lane & 15;
  const int rbase = (lane < 16) ? 0 : 8;
  const float bv = ba2[n];
#pragma unroll
  for (int j = 0; j < 8; ++j) sm[(rbase + j) * 16 + n] = acc[j] + bv;
  __syncthreads();
  if (lane < 16) {
    float mx = -3.0e38f;
#pragma unroll
    for (int i = 0; i < 16; ++i) mx = fmaxf(mx, sm[lane * 16 + i]);
    float e[16]; float s = 0.f;
#pragma unroll
    for (int i = 0; i < 16; ++i) { e[i] = __expf(sm[lane * 16 + i] - mx); s += e[i]; }
    const float inv = 1.f / s;
#pragma unroll
    for (int i = 0; i < 16; ++i) beta[(size_t)(m0 + lane) * RANK + i] = e[i] * inv;
  }
}

// ---------------------------------------------------------------------------
// Kernel 4: fused rank expansion + tri-linear product + beta reduction.
// z[b,:] = sum_r beta[b,r] * (g@Ur^T) * (d2@Vr^T) * (d3@Sr^T)
// 64 rows/block (128 blocks -> U/V/S L2 traffic 128 x 6 MB), 512 threads:
// wave owns a 16-col strip; per (rank, row-subtile): K=256 loop of 3 WMMAs;
// rank tensors never leave registers. LDS: 96 KB tiles + 4 KB beta.
// ---------------------------------------------------------------------------
__global__ __launch_bounds__(512)
void k_fuse(const bf16_t* __restrict__ gb, const bf16_t* __restrict__ d2b,
            const bf16_t* __restrict__ d3b, const bf16_t* __restrict__ Ub,
            const bf16_t* __restrict__ Vb, const bf16_t* __restrict__ Sb,
            const float* __restrict__ beta, float* __restrict__ out) {
  __shared__ bf16_t sG[64 * 256];
  __shared__ bf16_t s2[64 * 256];
  __shared__ bf16_t s3[64 * 256];
  __shared__ float  sBeta[64 * RANK];

  const int m0 = blockIdx.x * 64;
  const int t  = threadIdx.x;

  // Stage activation tiles (bf16) and beta tile into LDS.
  for (int c = t; c < 2048; c += 512) {   // 64*256/8 chunks of 8 bf16
    int row = c >> 5;
    int col = (c & 31) << 3;
    size_t go = (size_t)(m0 + row) * 256 + col;
    *(v8bf*)&sG[row * 256 + col] = *(const v8bf*)&gb[go];
    *(v8bf*)&s2[row * 256 + col] = *(const v8bf*)&d2b[go];
    *(v8bf*)&s3[row * 256 + col] = *(const v8bf*)&d3b[go];
  }
  for (int i = t; i < 64 * RANK; i += 512) sBeta[i] = beta[(size_t)m0 * RANK + i];
  __syncthreads();

  const int wave = t >> 5;
  const int lane = t & 31;
  const int nb   = wave * 16;

  v8f z[4] = {};

  for (int r = 0; r < RANK; ++r) {
    const bf16_t* Ur = Ub + (size_t)r * D_F * D_F;
    const bf16_t* Vr = Vb + (size_t)r * D_F * D_F;
    const bf16_t* Sr = Sb + (size_t)r * D_F * D_F;
    __builtin_prefetch(Ur + (size_t)nb * 256, 0, 1);

#pragma unroll
    for (int mt = 0; mt < 4; ++mt) {
      v8f ag = {}, a2acc = {}, a3acc = {};
      const bf16_t* Ag = sG + mt * 16 * 256;
      const bf16_t* A2 = s2 + mt * 16 * 256;
      const bf16_t* A3 = s3 + mt * 16 * 256;
#pragma unroll
      for (int k0 = 0; k0 < 256; k0 += 32) {
        v16bf aG = loadA_bf16(Ag, 256, 0, k0, lane);
        v16bf a2 = loadA_bf16(A2, 256, 0, k0, lane);
        v16bf a3 = loadA_bf16(A3, 256, 0, k0, lane);
        ag    = wmma_bf16(aG, loadB_bf16(Ur, 256, nb, k0, lane), ag);
        a2acc = wmma_bf16(a2, loadB_bf16(Vr, 256, nb, k0, lane), a2acc);
        a3acc = wmma_bf16(a3, loadB_bf16(Sr, 256, nb, k0, lane), a3acc);
      }
      const int rbase = mt * 16 + ((lane < 16) ? 0 : 8);
#pragma unroll
      for (int j = 0; j < 8; ++j) {
        float bw = sBeta[(rbase + j) * RANK + r];
        z[mt][j] += bw * ag[j] * a2acc[j] * a3acc[j];
      }
    }
  }

  // Store z (f32, B x 256)
  const int n0 = nb + (lane & 15);
#pragma unroll
  for (int mt = 0; mt < 4; ++mt) {
    const int rbase = mt * 16 + ((lane < 16) ? 0 : 8);
#pragma unroll
    for (int j = 0; j < 8; ++j)
      out[(size_t)(m0 + rbase + j) * 256 + n0] = z[mt][j];
  }
}

// ---------------------------------------------------------------------------
// Launch
// ---------------------------------------------------------------------------
extern "C" void kernel_launch(void* const* d_in, const int* in_sizes, int n_in,
                              void* d_out, int out_size, void* d_ws, size_t ws_size,
                              hipStream_t stream) {
  const float* h_g   = (const float*)d_in[0];
  const float* h_2d  = (const float*)d_in[1];
  const float* h_3d  = (const float*)d_in[2];
  const float* Wg    = (const float*)d_in[3];
  const float* bg    = (const float*)d_in[4];
  const float* W2    = (const float*)d_in[5];
  const float* b2    = (const float*)d_in[6];
  const float* W3    = (const float*)d_in[7];
  const float* b3    = (const float*)d_in[8];
  const float* ln_g_w = (const float*)d_in[9];
  const float* ln_g_b = (const float*)d_in[10];
  const float* ln_2_w = (const float*)d_in[11];
  const float* ln_2_b = (const float*)d_in[12];
  const float* ln_3_w = (const float*)d_in[13];
  const float* ln_3_b = (const float*)d_in[14];
  const float* U     = (const float*)d_in[15];
  const float* V     = (const float*)d_in[16];
  const float* S     = (const float*)d_in[17];
  const float* Wa1   = (const float*)d_in[18];
  const float* ba1   = (const float*)d_in[19];
  const float* Wa2   = (const float*)d_in[20];
  const float* ba2   = (const float*)d_in[21];

  // Workspace carve-up (bf16 stages), ~30 MB total.
  char* ws = (char*)d_ws;
  bf16_t* Ub     = (bf16_t*)ws; ws += (size_t)RANK * D_F * D_F * sizeof(bf16_t); // 2 MB
  bf16_t* Vb     = (bf16_t*)ws; ws += (size_t)RANK * D_F * D_F * sizeof(bf16_t);
  bf16_t* Sb     = (bf16_t*)ws; ws += (size_t)RANK * D_F * D_F * sizeof(bf16_t);
  bf16_t* gbuf   = (bf16_t*)ws; ws += (size_t)BATCH * D_F * sizeof(bf16_t);      // 4 MB
  bf16_t* d2buf  = (bf16_t*)ws; ws += (size_t)BATCH * D_F * sizeof(bf16_t);
  bf16_t* d3buf  = (bf16_t*)ws; ws += (size_t)BATCH * D_F * sizeof(bf16_t);
  bf16_t* hidden = (bf16_t*)ws; ws += (size_t)BATCH * ATTN_H * sizeof(bf16_t);   // 8 MB
  float*  betab  = (float*)ws;  ws += (size_t)BATCH * RANK * sizeof(float);      // 0.5 MB
  bf16_t* Wgb    = (bf16_t*)ws; ws += (size_t)D_F * D_G  * sizeof(bf16_t);
  bf16_t* W2b    = (bf16_t*)ws; ws += (size_t)D_F * D_2D * sizeof(bf16_t);
  bf16_t* W3b    = (bf16_t*)ws; ws += (size_t)D_F * D_3D * sizeof(bf16_t);
  bf16_t* Wa1b   = (bf16_t*)ws; ws += (size_t)ATTN_H * D_CAT * sizeof(bf16_t);   // 2.25 MB
  bf16_t* Wa2b   = (bf16_t*)ws; ws += (size_t)RANK * ATTN_H * sizeof(bf16_t);

  // One-time f32 -> bf16 weight conversions.
  const int nUVS = RANK * D_F * D_F;
  k_cvt<<<(nUVS + 255) / 256, 256, 0, stream>>>(U, Ub, nUVS);
  k_cvt<<<(nUVS + 255) / 256, 256, 0, stream>>>(V, Vb, nUVS);
  k_cvt<<<(nUVS + 255) / 256, 256, 0, stream>>>(S, Sb, nUVS);
  k_cvt<<<(D_F * D_G  + 255) / 256, 256, 0, stream>>>(Wg,  Wgb,  D_F * D_G);
  k_cvt<<<(D_F * D_2D + 255) / 256, 256, 0, stream>>>(W2,  W2b,  D_F * D_2D);
  k_cvt<<<(D_F * D_3D + 255) / 256, 256, 0, stream>>>(W3,  W3b,  D_F * D_3D);
  k_cvt<<<(ATTN_H * D_CAT + 255) / 256, 256, 0, stream>>>(Wa1, Wa1b, ATTN_H * D_CAT);
  k_cvt<<<(RANK * ATTN_H + 255) / 256, 256, 0, stream>>>(Wa2, Wa2b, RANK * ATTN_H);

  k_proj_ln_relu<<<BATCH / 64, 512, 0, stream>>>(h_g,  Wgb, bg, ln_g_w, ln_g_b, gbuf,  D_G);
  k_proj_ln_relu<<<BATCH / 64, 512, 0, stream>>>(h_2d, W2b, b2, ln_2_w, ln_2_b, d2buf, D_2D);
  k_proj_ln_relu<<<BATCH / 64, 512, 0, stream>>>(h_3d, W3b, b3, ln_3_w, ln_3_b, d3buf, D_3D);

  k_attn1<<<dim3(BATCH / 64, 2), 512, 0, stream>>>(h_g, h_2d, h_3d, Wa1b, ba1, hidden);
  k_attn2_softmax<<<BATCH / 16, 32, 0, stream>>>(hidden, Wa2b, ba2, betab);

  k_fuse<<<BATCH / 64, 512, 0, stream>>>(gbuf, d2buf, d3buf, Ub, Vb, Sb, betab, (float*)d_out);
}
